// GCNGuard_18047452578191
// MI455X (gfx1250) — compile-verified, matrix operand order checked by reading, not run on hardware
//
#include <hip/hip_runtime.h>
#include <math.h>

// ---------------------------------------------------------------------------
// GCNGuard 2-layer forward for MI455X (gfx1250, wave32).
// Dense projections use v_wmma_f32_16x16x32_bf16; graph ops use wave-per-edge
// coalesced gathers + global f32 atomics (feature matrix is L2-resident).
// ---------------------------------------------------------------------------

typedef __attribute__((ext_vector_type(16))) __bf16        v16bf;
typedef __attribute__((ext_vector_type(8)))  float         v8f;
typedef __attribute__((ext_vector_type(4)))  float         vf4;
typedef __attribute__((ext_vector_type(4)))  unsigned int  vu4;

#define EPS_F    1e-8f
#define LN_EPS_F 1e-5f

__device__ __forceinline__ float wave_sum(float v) {
#pragma unroll
  for (int m = 16; m >= 1; m >>= 1) v += __shfl_xor(v, m, 32);
  return v;
}
__device__ __forceinline__ float wave_max(float v) {
#pragma unroll
  for (int m = 16; m >= 1; m >>= 1) v = fmaxf(v, __shfl_xor(v, m, 32));
  return v;
}
__device__ __forceinline__ unsigned short f2bf(float f) {
  unsigned int u = __float_as_uint(f);
  u += 0x7FFFu + ((u >> 16) & 1u);             // round-to-nearest-even
  return (unsigned short)(u >> 16);
}

// ---- elementwise utilities -------------------------------------------------

__global__ void k_fill(float* __restrict__ p, float v, long n) {
  long i = blockIdx.x * (long)blockDim.x + threadIdx.x;
  if (i < n) p[i] = v;
}

__global__ void k_f32_to_bf16(const float* __restrict__ s, unsigned short* __restrict__ d, long n) {
  long i = blockIdx.x * (long)blockDim.x + threadIdx.x;
  if (i < n) d[i] = f2bf(s[i]);
}

// Pack W (K x DN, f32 row-major) into per-lane-contiguous bf16 WMMA B-fragments:
// P[((nt*ksteps + ks)*32 + lane)*16 + v*2 + h] = bf16(W[(ks*32+k)*DN + nt*16 + n])
// with k = (v%4)*2 + (v/4)*16 + (lane/16)*8 + h, n = lane%16 (ISA 16-bit layout).
__global__ void k_pack_w(const float* __restrict__ W, unsigned short* __restrict__ P,
                         int K, int DN) {
  int idx = blockIdx.x * blockDim.x + threadIdx.x;
  int ksteps = K >> 5;
  int total = (DN >> 4) * ksteps * 512;
  if (idx >= total) return;
  int h    = idx & 1;
  int v    = (idx >> 1) & 7;
  int lane = (idx >> 4) & 31;
  int ks   = (idx >> 9) % ksteps;
  int nt   = (idx >> 9) / ksteps;
  int k = ((v & 3) << 1) + ((v >> 2) << 4) + ((lane >> 4) << 3) + h;
  int n = lane & 15;
  P[idx] = f2bf(W[(size_t)(ks * 32 + k) * DN + nt * 16 + n]);
}

// ---- attention (att_coef) --------------------------------------------------

// wave per node: nrm[i] = max(||F_i||2, EPS)  (D = 128)
__global__ void k_row_norm(const float* __restrict__ F, float* __restrict__ nrm, int n) {
  long t = blockIdx.x * (long)blockDim.x + threadIdx.x;
  int lane = threadIdx.x & 31;
  long node = t >> 5;
  if (node >= n) return;
  vf4 a = *(const vf4*)(F + node * 128 + lane * 4);
  float s = wave_sum(a.x * a.x + a.y * a.y + a.z * a.z + a.w * a.w);
  if (lane == 0) nrm[node] = fmaxf(sqrtf(s), EPS_F);
}

// wave per edge: cosine sim, threshold, segment sums (degA by row, cnt by col)
__global__ void k_edge_att(const float* __restrict__ F, const int* __restrict__ row,
                           const int* __restrict__ col, const float* __restrict__ nrm,
                           float* __restrict__ sims, float* __restrict__ degA,
                           float* __restrict__ cnt, long E) {
  long t = blockIdx.x * (long)blockDim.x + threadIdx.x;
  int lane = threadIdx.x & 31;
  long e = t >> 5;
  if (e >= E) return;
  int r = row[e], c = col[e];
  vf4 a = *(const vf4*)(F + (size_t)r * 128 + lane * 4);
  vf4 b = *(const vf4*)(F + (size_t)c * 128 + lane * 4);
  float d = wave_sum(a.x * b.x + a.y * b.y + a.z * b.z + a.w * b.w);
  if (lane == 0) {
    float s = d / (nrm[r] * nrm[c]);
    if (s >= 0.1f) {
      sims[e] = s;
      atomicAdd(&degA[r], s);
      atomicAdd(&cnt[c], 1.0f);
    } else {
      sims[e] = 0.0f;
    }
  }
}

// per node: dinvA, self-loop weight exp(1/(cnt+1)); seed conv-degree with it
__global__ void k_node1(const float* __restrict__ degA, const float* __restrict__ cnt,
                        float* __restrict__ dinvA, float* __restrict__ sexp,
                        float* __restrict__ deg2, int n) {
  int i = blockIdx.x * blockDim.x + threadIdx.x;
  if (i >= n) return;
  float dA = degA[i];
  dinvA[i] = (dA > 0.0f) ? rsqrtf(dA) : 0.0f;
  float se = __expf(1.0f / (cnt[i] + 1.0f));
  sexp[i] = se;
  deg2[i] = se;                      // self-loop term of segsum(w, col)
}

// per edge: vals = exp(dinvA[r]*sims*dinvA[c]) for kept edges; deg2[col] += vals
__global__ void k_edge_val(const int* __restrict__ row, const int* __restrict__ col,
                           const float* __restrict__ sims, const float* __restrict__ dinvA,
                           float* __restrict__ vals, float* __restrict__ deg2, long E) {
  long e = blockIdx.x * (long)blockDim.x + threadIdx.x;
  if (e >= E) return;
  float s = sims[e];
  float v = 0.0f;
  if (s > 0.0f) {
    v = __expf(dinvA[row[e]] * s * dinvA[col[e]]);
    atomicAdd(&deg2[col[e]], v);
  }
  vals[e] = v;
}

__global__ void k_node2(const float* __restrict__ deg2, float* __restrict__ dinv2, int n) {
  int i = blockIdx.x * blockDim.x + threadIdx.x;
  if (i >= n) return;
  float d = deg2[i];
  dinv2[i] = (d > 0.0f) ? rsqrtf(d) : 0.0f;
}

// ---- dense projection: C[M x DN] = A[M x K](bf16) @ Wpack (WMMA) -----------

__global__ void k_gemm_wmma(const unsigned short* __restrict__ A,
                            const unsigned short* __restrict__ Bp,
                            float* __restrict__ C, int M, int K, int DN) {
  int lane = threadIdx.x & 31;
  int wave = threadIdx.x >> 5;
  int mt = blockIdx.x;
  int nt = blockIdx.y * 4 + wave;             // 4 waves per block, one 16x16 tile each
  if (nt * 16 >= DN) return;                  // wave-uniform
  int ksteps = K >> 5;
  int m = lane & 15;
  int rowi = mt * 16 + m;
  int rowc = (rowi < M) ? rowi : (M - 1);     // clamp loads; stores predicated
  // A fragment (16x32 bf16): per-lane halves are two contiguous 16B chunks.
  const unsigned short* pa = A + (size_t)rowc * K + ((lane >> 4) << 3);
  const unsigned short* pb = Bp + ((size_t)nt * ksteps * 32 + lane) * 16;
  v8f acc = {0.f, 0.f, 0.f, 0.f, 0.f, 0.f, 0.f, 0.f};
  for (int ks = 0; ks < ksteps; ++ks) {
    union { vu4 q[2]; v16bf h; } fa, fb;
    fa.q[0] = *(const vu4*)(pa);
    fa.q[1] = *(const vu4*)(pa + 16);
    fb.q[0] = *(const vu4*)(pb);
    fb.q[1] = *(const vu4*)(pb + 8);
    acc = __builtin_amdgcn_wmma_f32_16x16x32_bf16(false, fa.h, false, fb.h,
                                                  (short)0, acc, false, false);
    pa += 32;          // next K=32 chunk
    pb += 512;         // next packed fragment block (32 lanes * 16 halves)
  }
  int mb = (lane >> 4) << 3;                  // C layout: VGPR r -> M = mb + r
  float* cp = C + (size_t)(mt * 16) * DN + nt * 16 + m;
#pragma unroll
  for (int r = 0; r < 8; ++r) {
    int mm = mt * 16 + mb + r;
    if (mm < M) cp[(size_t)(mb + r) * DN] = acc[r];
  }
}

// ---- GCN conv: self-loop init + edge scatter -------------------------------

__global__ void k_out_init(const float* __restrict__ gem, const float* __restrict__ bias,
                           const float* __restrict__ dinv2, const float* __restrict__ sexp,
                           float* __restrict__ out, int n, int DN) {
  long idx = blockIdx.x * (long)blockDim.x + threadIdx.x;
  if (idx >= (long)n * DN) return;
  int i = (int)(idx / DN);
  int d = (int)(idx % DN);
  float di = dinv2[i];
  out[idx] = di * di * sexp[i] * gem[idx] + bias[d];
}

__global__ void k_scatter(const int* __restrict__ row, const int* __restrict__ col,
                          const float* __restrict__ vals, const float* __restrict__ dinv2,
                          const float* __restrict__ gem, float* __restrict__ out,
                          long E, int DN) {
  long t = blockIdx.x * (long)blockDim.x + threadIdx.x;
  int lane = threadIdx.x & 31;
  long e = t >> 5;
  if (e >= E) return;
  float v = vals[e];
  if (v == 0.0f) return;                      // wave-uniform (filtered edge)
  int r = row[e], c = col[e];
  float coef = dinv2[r] * v * dinv2[c];
  const float* src = gem + (size_t)r * DN;
  float* dst = out + (size_t)c * DN;
  for (int d = lane; d < DN; d += 32)
    atomicAdd(&dst[d], coef * src[d]);
}

// ---- post ops --------------------------------------------------------------

// wave per node: LayerNorm(128) + ReLU; emit f32 (attention) + bf16 (next GEMM)
__global__ void k_ln_relu(const float* __restrict__ in, const float* __restrict__ g,
                          const float* __restrict__ b, float* __restrict__ outf,
                          unsigned short* __restrict__ outbf, int n) {
  long t = blockIdx.x * (long)blockDim.x + threadIdx.x;
  int lane = threadIdx.x & 31;
  long node = t >> 5;
  if (node >= n) return;
  vf4 x = *(const vf4*)(in + node * 128 + lane * 4);
  float mu = wave_sum(x.x + x.y + x.z + x.w) * (1.0f / 128.0f);
  vf4 xm = x - mu;
  float var = wave_sum(xm.x * xm.x + xm.y * xm.y + xm.z * xm.z + xm.w * xm.w) * (1.0f / 128.0f);
  float rs = rsqrtf(var + LN_EPS_F);
  vf4 gg = *(const vf4*)(g + lane * 4);
  vf4 bb = *(const vf4*)(b + lane * 4);
  vf4 y = xm * rs * gg + bb;
  y.x = fmaxf(y.x, 0.0f); y.y = fmaxf(y.y, 0.0f);
  y.z = fmaxf(y.z, 0.0f); y.w = fmaxf(y.w, 0.0f);
  *(vf4*)(outf + node * 128 + lane * 4) = y;
  unsigned short* ob = outbf + node * 128 + lane * 4;
  ob[0] = f2bf(y.x); ob[1] = f2bf(y.y); ob[2] = f2bf(y.z); ob[3] = f2bf(y.w);
}

// wave per node: in-place log_softmax over 64 dims (2 per lane)
__global__ void k_logsoftmax(float* __restrict__ out, int n) {
  long t = blockIdx.x * (long)blockDim.x + threadIdx.x;
  int lane = threadIdx.x & 31;
  long node = t >> 5;
  if (node >= n) return;
  float* p = out + node * 64;
  float x0 = p[lane], x1 = p[lane + 32];
  float mx = wave_max(fmaxf(x0, x1));
  float se = wave_sum(__expf(x0 - mx) + __expf(x1 - mx));
  float lse = mx + __logf(se);
  p[lane] = x0 - lse;
  p[lane + 32] = x1 - lse;
}

// ---------------------------------------------------------------------------

static inline int cdivl(long a, long b) { return (int)((a + b - 1) / b); }

static void run_layer(const float* feats, const unsigned short* featsbf,
                      const unsigned short* wpack, const float* bias,
                      float* outbuf, int N, long E, int K, int DN,
                      const int* row, const int* col,
                      float* nrm, float* degA, float* cnt, float* sexp,
                      float* dinvA, float* deg2, float* dinv2,
                      float* sims, float* vals, float* gem, hipStream_t stream) {
  const int BT = 256;
  const int nodeBlk = cdivl((long)N, BT);
  const int nodeWaveBlk = cdivl((long)N * 32, BT);
  const int edgeWaveBlk = cdivl(E * 32, BT);
  // attention
  k_row_norm<<<nodeWaveBlk, BT, 0, stream>>>(feats, nrm, N);
  k_fill<<<nodeBlk, BT, 0, stream>>>(degA, 0.0f, N);
  k_fill<<<nodeBlk, BT, 0, stream>>>(cnt, 0.0f, N);
  k_edge_att<<<edgeWaveBlk, BT, 0, stream>>>(feats, row, col, nrm, sims, degA, cnt, E);
  k_node1<<<nodeBlk, BT, 0, stream>>>(degA, cnt, dinvA, sexp, deg2, N);
  k_edge_val<<<cdivl(E, BT), BT, 0, stream>>>(row, col, sims, dinvA, vals, deg2, E);
  k_node2<<<nodeBlk, BT, 0, stream>>>(deg2, dinv2, N);
  // dense projection (WMMA)
  dim3 gg(cdivl((long)N, 16), cdivl((long)DN, 64));
  k_gemm_wmma<<<gg, 128, 0, stream>>>(featsbf, wpack, gem, N, K, DN);
  // conv: self-loops then edge scatter
  k_out_init<<<cdivl((long)N * DN, BT), BT, 0, stream>>>(gem, bias, dinv2, sexp, outbuf, N, DN);
  k_scatter<<<edgeWaveBlk, BT, 0, stream>>>(row, col, vals, dinv2, gem, outbuf, E, DN);
}

extern "C" void kernel_launch(void* const* d_in, const int* in_sizes, int n_in,
                              void* d_out, int out_size, void* d_ws, size_t ws_size,
                              hipStream_t stream) {
  const float* x   = (const float*)d_in[0];
  const int*   row = (const int*)d_in[1];
  const int*   col = (const int*)d_in[2];
  const float* W1  = (const float*)d_in[3];
  const float* b1  = (const float*)d_in[4];
  const float* lng = (const float*)d_in[5];
  const float* lnb = (const float*)d_in[6];
  const float* W2  = (const float*)d_in[7];
  const float* b2  = (const float*)d_in[8];
  (void)n_in; (void)out_size; (void)ws_size;

  const int DIN = 128, DH = 128, DOUT = 64;
  const int  N = in_sizes[0] / DIN;
  const long E = in_sizes[1];
  float* out = (float*)d_out;

  // workspace carve-out (256B aligned slices)
  char* base = (char*)d_ws;
  size_t off = 0;
  auto take = [&](size_t bytes) -> void* {
    void* p = base + off;
    off += (bytes + 255) & ~(size_t)255;
    return p;
  };
  unsigned short* xbf  = (unsigned short*)take((size_t)N * DIN * 2);
  unsigned short* hbf  = (unsigned short*)take((size_t)N * DH * 2);
  float* hfe   = (float*)take((size_t)N * DH * 4);   // post LN+ReLU features
  float* gem   = (float*)take((size_t)N * DH * 4);   // GEMM output
  float* out1  = (float*)take((size_t)N * DH * 4);   // layer-1 conv accum
  float* nrm   = (float*)take((size_t)N * 4);
  float* degA  = (float*)take((size_t)N * 4);
  float* cnt   = (float*)take((size_t)N * 4);
  float* sexp  = (float*)take((size_t)N * 4);
  float* dinvA = (float*)take((size_t)N * 4);
  float* deg2  = (float*)take((size_t)N * 4);
  float* dinv2 = (float*)take((size_t)N * 4);
  float* sims  = (float*)take((size_t)E * 4);
  float* vals  = (float*)take((size_t)E * 4);
  unsigned short* wp1 = (unsigned short*)take((size_t)(DH / 16) * (DIN / 32) * 512 * 2);
  unsigned short* wp2 = (unsigned short*)take((size_t)(DOUT / 16) * (DH / 32) * 512 * 2);

  const int BT = 256;
  // precision prep: bf16 activations + packed bf16 WMMA B-fragments
  k_f32_to_bf16<<<cdivl((long)N * DIN, BT), BT, 0, stream>>>(x, xbf, (long)N * DIN);
  k_pack_w<<<cdivl((long)(DH / 16) * (DIN / 32) * 512, BT), BT, 0, stream>>>(W1, wp1, DIN, DH);
  k_pack_w<<<cdivl((long)(DOUT / 16) * (DH / 32) * 512, BT), BT, 0, stream>>>(W2, wp2, DH, DOUT);

  // layer 1: attention on x -> conv (W1) -> LN+ReLU
  run_layer(x, xbf, wp1, b1, out1, N, E, DIN, DH, row, col,
            nrm, degA, cnt, sexp, dinvA, deg2, dinv2, sims, vals, gem, stream);
  k_ln_relu<<<cdivl((long)N * 32, BT), BT, 0, stream>>>(out1, lng, lnb, hfe, hbf, N);

  // layer 2: attention on h -> conv (W2) -> log_softmax (in-place on d_out)
  run_layer(hfe, hbf, wp2, b2, out, N, E, DH, DOUT, row, col,
            nrm, degA, cnt, sexp, dinvA, deg2, dinv2, sims, vals, gem, stream);
  k_logsoftmax<<<cdivl((long)N * 32, BT), BT, 0, stream>>>(out, N);
}
